// fkan_layer_91044716741313
// MI455X (gfx1250) — matrix operand order repaired and checked by reading, not compiled, not run on hardware
//
#include <hip/hip_runtime.h>
#include <hip/hip_bf16.h>
#include <type_traits>

// Fourier-KAN layer as an f16 WMMA GEMM on gfx1250 (wave32).
//   y[b,o] = sum_{i,k} sin(k*x[b,i]) a[o,i,k] + cos(k*x[b,i]) b[o,i,k]
// == Y(4096x1024) = S(4096x16384) * W(1024x16384)^T
// Compute-bound (137 GFLOP vs ~100MB HBM; weights L2-resident) -> all math
// through v_wmma_f32_16x16x32_f16. Weight tiles staged memory->LDS via the
// CDNA5 async path (global_load_async_to_lds_b128, ASYNCcnt). The K loop is
// split into a sin phase and a cos phase so harmonic evaluation is branch-free,
// and x loads are issued before the WMMA block so XDL latency hides them.

#define IN_F      1024
#define OUT_F     1024
#define KHARM     8
#define BATCH     4096
#define KTOT      (2 * IN_F * KHARM)   // 16384 reduction length
#define HALF_K    (IN_F * KHARM)       // 8192 (sin | cos split)

#define BM        128                  // batch rows per block
#define BN        128                  // out features per block
#define BK        32                   // K slice per stage (1 wmma k-chunk)
#define PITCH     40                   // LDS row pitch in halfs: 80B = 20 banks
                                       // -> b128 frag reads touch all 64 banks
                                       //    exactly once per 16 lanes (optimal)

typedef __attribute__((ext_vector_type(16))) _Float16 v16h;
typedef __attribute__((ext_vector_type(8)))  float    v8f;
typedef __attribute__((ext_vector_type(4)))  int      v4i;

union Frag16 { v16h h; uint4 q[2]; };
union Pack8  { _Float16 h[8]; uint4 q; };

#if __has_builtin(__builtin_amdgcn_global_load_async_to_lds_b128)
#define HAS_ASYNC_LDS 1
// builtin prototype (from compiler diagnostic): (v4i AS1*, v4i AS3*, Imm, Imm)
typedef __attribute__((address_space(1))) v4i* gptr_v4i;
typedef __attribute__((address_space(3))) v4i* lptr_v4i;
#else
#define HAS_ASYNC_LDS 0
#endif

__device__ __forceinline__ void wait_asynccnt0() {
#if HAS_ASYNC_LDS
#if __has_builtin(__builtin_amdgcn_s_wait_asynccnt)
    __builtin_amdgcn_s_wait_asynccnt(0);
#else
    asm volatile("s_wait_asynccnt 0" ::: "memory");
#endif
#endif
}

using SinTag = std::integral_constant<bool, false>;
using CosTag = std::integral_constant<bool, true>;

// ---------------------------------------------------------------------------
// Kernel 1: pack [a | b] (fp32) into f16 weight matrix W16[OUT_F][KTOT] in ws.
// ---------------------------------------------------------------------------
__global__ __launch_bounds__(256)
void fkan_prep_w(const float* __restrict__ a, const float* __restrict__ b,
                 _Float16* __restrict__ w16) {
    size_t base = ((size_t)blockIdx.x * blockDim.x + threadIdx.x) * 8;
    if (base >= (size_t)OUT_F * KTOT) return;
    int n = (int)(base >> 14);          // / KTOT
    int c = (int)(base & (KTOT - 1));
    const float* src = (c < HALF_K) ? (a + (size_t)n * HALF_K + c)
                                    : (b + (size_t)n * HALF_K + (c - HALF_K));
    float4 v0 = ((const float4*)src)[0];
    float4 v1 = ((const float4*)src)[1];
    Pack8 pk;
    pk.h[0] = (_Float16)v0.x; pk.h[1] = (_Float16)v0.y;
    pk.h[2] = (_Float16)v0.z; pk.h[3] = (_Float16)v0.w;
    pk.h[4] = (_Float16)v1.x; pk.h[5] = (_Float16)v1.y;
    pk.h[6] = (_Float16)v1.z; pk.h[7] = (_Float16)v1.w;
    *(uint4*)(w16 + base) = pk.q;
}

// ---------------------------------------------------------------------------
// Kernel 2: fused feature-build + WMMA GEMM.
// Block: 256 threads = 8 waves, tile 128x128, double-buffered LDS (40KB).
// Wave grid 4(M) x 2(N); each wave owns a 32x64 patch = 2x4 16x16 accums.
// ---------------------------------------------------------------------------
__global__ __launch_bounds__(256)
void fkan_gemm(const float* __restrict__ x, const _Float16* __restrict__ w16,
               float* __restrict__ y) {
    __shared__ _Float16 sA[2][BM][PITCH];   // sin/cos features (f16)
    __shared__ _Float16 sB[2][BN][PITCH];   // packed weights  (f16)

    const int tid   = threadIdx.x;
    const int lane  = tid & 31;
    const int wave  = tid >> 5;
    const int wm    = wave & 3;             // 4 wave rows (M), 32 rows each
    const int wn    = wave >> 2;            // 2 wave cols (N), 64 cols each
    const int m0    = blockIdx.x * BM;      // gridDim.x = BATCH/BM = 32
    const int n0    = blockIdx.y * BN;      // gridDim.y = OUT_F/BN = 8

    v8f acc[2][4] = {};

    // staging maps (256 threads, 16 halfs = 32B each)
    const int srow = tid >> 1;              // 0..127 (row for both tiles)
    const int scol = (tid & 1) << 4;        // 0 / 16 halfs
    const float* xrow = x + (size_t)(m0 + srow) * IN_F;

    // x values for the two features this thread stages (consecutive, 8B aligned)
    auto load_x2 = [&](int cbase) -> float2 {
        return *(const float2*)(xrow + ((cbase + scol) >> 3));
    };

    // Issue async memory->LDS copy of a 128x32 f16 weight tile (no VGPR data).
    auto stage_B = [&](int buf, int ctot) {
        const _Float16* gsrc = w16 + (size_t)(n0 + srow) * KTOT + ctot + scol;
        _Float16*       ldst = &sB[buf][srow][scol];
#if HAS_ASYNC_LDS
        __builtin_amdgcn_global_load_async_to_lds_b128(
            (gptr_v4i)gsrc,       (lptr_v4i)ldst,       0, 0);
        __builtin_amdgcn_global_load_async_to_lds_b128(
            (gptr_v4i)(gsrc + 8), (lptr_v4i)(ldst + 8), 0, 0);
#else
        ((uint4*)ldst)[0] = ((const uint4*)gsrc)[0];
        ((uint4*)ldst)[1] = ((const uint4*)gsrc)[1];
#endif
    };

    // Branch-free harmonic evaluation: IS_COS is a compile-time constant.
    auto stage_A = [&](int buf, float2 xv, auto ic) {
        constexpr bool IS_COS = decltype(ic)::value;
#pragma unroll
        for (int f = 0; f < 2; ++f) {
            const float xvf = (f == 0) ? xv.x : xv.y;
            Pack8 pk;
#pragma unroll
            for (int k = 0; k < 8; ++k) {
                float ph = xvf * (float)k;
                float v  = IS_COS ? __cosf(ph) : __sinf(ph);
                pk.h[k]  = (_Float16)v;
            }
            *(uint4*)&sA[buf][srow][scol + f * 8] = pk.q;
        }
    };

    auto compute = [&](int buf) {
        const int r16 = lane & 15;
        const int ahk = (lane < 16) ? 0 : 8;    // A-layout K sub-chunk
        const int bhk = (lane < 16) ? 0 : 8;    // B-layout K sub-chunk
        Frag16 af[2], bf[4];
#pragma unroll
        for (int mt = 0; mt < 2; ++mt) {
            const _Float16* p = &sA[buf][wm * 32 + mt * 16 + r16][ahk];
            af[mt].q[0] = *(const uint4*)(p);        // K 0-7  / 8-15
            af[mt].q[1] = *(const uint4*)(p + 16);   // K16-23 / 24-31
        }
#pragma unroll
        for (int nt = 0; nt < 4; ++nt) {
            const _Float16* p = &sB[buf][wn * 64 + nt * 16 + r16][bhk];
            bf[nt].q[0] = *(const uint4*)(p);        // K 0-7  / 8-15
            bf[nt].q[1] = *(const uint4*)(p + 16);   // K16-23 / 24-31
        }
#pragma unroll
        for (int mt = 0; mt < 2; ++mt)
#pragma unroll
            for (int nt = 0; nt < 4; ++nt)
                acc[mt][nt] = __builtin_amdgcn_wmma_f32_16x16x32_f16(
                    false, af[mt].h, false, bf[nt].h,
                    (short)0, acc[mt][nt], false, false);
    };

    // ---- software pipeline over 512 K-steps: [0,256) sin | [256,512) cos ----
    // Per iteration: issue x load + async B for tile s+1, run WMMAs on tile s,
    // then do sin/cos math (x load latency hidden behind the XDL block).
    const int HSTEPS = HALF_K / BK;         // 256

    {   // prologue: stage sin tile 0
        float2 xv = load_x2(0);
        stage_B(0, 0);
        stage_A(0, xv, SinTag{});
        wait_asynccnt0();
        __syncthreads();
    }
    for (int s = 0; s < HSTEPS - 1; ++s) {  // sin phase, stages sin tiles 1..255
        const int nc = (s + 1) * BK;
        float2 xn = load_x2(nc);
        stage_B((s + 1) & 1, nc);
        compute(s & 1);
        stage_A((s + 1) & 1, xn, SinTag{});
        wait_asynccnt0();
        __syncthreads();
    }
    {   // boundary: compute last sin tile (buf 1), stage cos tile 0 (buf 0)
        float2 xn = load_x2(0);
        stage_B(0, HALF_K);
        compute(1);
        stage_A(0, xn, CosTag{});
        wait_asynccnt0();
        __syncthreads();
    }
    for (int s = HSTEPS; s < 2 * HSTEPS - 1; ++s) {  // cos phase
        const int nc = (s + 1 - HSTEPS) * BK;
        float2 xn = load_x2(nc);
        stage_B((s + 1) & 1, HALF_K + nc);
        compute(s & 1);
        stage_A((s + 1) & 1, xn, CosTag{});
        wait_asynccnt0();
        __syncthreads();
    }
    compute(1);                              // last cos tile

    // Epilogue: C/D layout -> VGPR j holds M = j (+8 for lanes 16-31), N = lane%16
    const int mbase = m0 + wm * 32 + ((lane < 16) ? 0 : 8);
    const int nlane = lane & 15;
#pragma unroll
    for (int mt = 0; mt < 2; ++mt) {
#pragma unroll
        for (int nt = 0; nt < 4; ++nt) {
            const int ncol = n0 + wn * 64 + nt * 16 + nlane;
#pragma unroll
            for (int j = 0; j < 8; ++j)
                y[(size_t)(mbase + mt * 16 + j) * OUT_F + ncol] = acc[mt][nt][j];
        }
    }
}

// ---------------------------------------------------------------------------
extern "C" void kernel_launch(void* const* d_in, const int* in_sizes, int n_in,
                              void* d_out, int out_size, void* d_ws, size_t ws_size,
                              hipStream_t stream) {
    (void)in_sizes; (void)n_in; (void)out_size; (void)ws_size;
    const float* x = (const float*)d_in[0];   // [4096, 1024]
    const float* a = (const float*)d_in[1];   // [1024, 1024, 8]
    const float* b = (const float*)d_in[2];   // [1024, 1024, 8]
    float*       y = (float*)d_out;           // [4096, 1024]
    _Float16*  w16 = (_Float16*)d_ws;         // 32 MB packed f16 weights

    // 1) pack weights to f16 (16M elems / 8 per thread)
    fkan_prep_w<<<(OUT_F * KTOT) / (256 * 8), 256, 0, stream>>>(a, b, w16);

    // 2) fused sin/cos + WMMA GEMM
    dim3 grid(BATCH / BM, OUT_F / BN);        // 32 x 8
    fkan_gemm<<<grid, 256, 0, stream>>>(x, w16, y);
}
